// QuaNet_link_prediction_one_laplacian_44573170598898
// MI455X (gfx1250) — matrix operand order repaired and checked by reading, not compiled, not run on hardware
//
#include <hip/hip_runtime.h>
#include <hip/hip_bf16.h>

typedef __attribute__((ext_vector_type(16))) _Float16 v16h;
typedef __attribute__((ext_vector_type(8)))  _Float16 v8h;
typedef __attribute__((ext_vector_type(8)))  float    v8f;

constexpr int N_NODES = 100000;
constexpr int N_EDGES = 1000000;
constexpr int N_QUERY = 100000;
constexpr int KDIM    = 256;   // 4 quaternion components x 64 features, concatenated

// ---------------------------------------------------------------------------
// Zero-fill (T accumulator must be zeroed before each scatter pass)
// ---------------------------------------------------------------------------
__global__ void zero_f32(float* __restrict__ p, int n4) {
  int i = blockIdx.x * blockDim.x + threadIdx.x;
  if (i < n4) ((float4*)p)[i] = make_float4(0.f, 0.f, 0.f, 0.f);
}

// ---------------------------------------------------------------------------
// Build W_big (256x256, f16) directly in the WMMA B-fragment-swizzled layout:
//   flat index = ((kc*16 + ntile)*32 + lane)*16 + e
//   K = kc*32 + (lane>=16 ? 16 : 0) + e ;  N = ntile*16 + (lane&15)
// Block (a,c) of W_big is sign[a][c] * W[src[a][c]]  (quaternion matmul).
// ---------------------------------------------------------------------------
__global__ void build_wbig(const float* __restrict__ W, _Float16* __restrict__ Wswz) {
  int idx  = blockIdx.x * 256 + threadIdx.x;      // 65536 total
  int e    = idx & 15;
  int lane = (idx >> 4) & 31;
  int nt   = (idx >> 9) & 15;
  int kc   = idx >> 13;
  int K  = kc * 32 + ((lane >> 4) << 4) + e;
  int Nc = nt * 16 + (lane & 15);
  int a = K >> 6, d = K & 63;
  int c = Nc >> 6, h = Nc & 63;
  const int   SRCT[4][4] = {{0,1,2,3},{1,0,3,2},{2,3,0,1},{3,2,1,0}};
  const float SGNT[4][4] = {{ 1.f, 1.f, 1.f, 1.f},
                            {-1.f, 1.f,-1.f, 1.f},
                            {-1.f, 1.f, 1.f,-1.f},
                            {-1.f,-1.f, 1.f, 1.f}};
  float v = SGNT[a][c] * W[(SRCT[a][c] * 64 + d) * 64 + h];
  Wswz[idx] = (_Float16)v;
}

// ---------------------------------------------------------------------------
// Edge aggregation: per edge, quaternion product m (x) G[src] scattered to dst.
// 64 threads per edge (one per feature), 4 edges per 256-thread block.
// Layer 1: g* = Xr/Xi/Xj/Xk (stride 64); Layer 2: g* = H+a*64 (stride 256).
// ---------------------------------------------------------------------------
__global__ void edge_scatter(const float* __restrict__ g0, const float* __restrict__ g1,
                             const float* __restrict__ g2, const float* __restrict__ g3,
                             int gstride,
                             const int* __restrict__ ei,
                             const float* __restrict__ nr, const float* __restrict__ ni,
                             const float* __restrict__ nj, const float* __restrict__ nk,
                             float* __restrict__ T) {
  int e = blockIdx.x * 4 + (threadIdx.x >> 6);
  int h = threadIdx.x & 63;
  if (e >= N_EDGES) return;
  int src = ei[e];
  int dst = ei[N_EDGES + e];
  float mr = -nr[e], mi = -ni[e], mj = -nj[e], mk = -nk[e];
  long long b = (long long)src * gstride;
  float a0 = g0[b + h], a1 = g1[b + h], a2 = g2[b + h], a3 = g3[b + h];
  float tr = mr*a0 - mi*a1 - mj*a2 - mk*a3;
  float ti = mr*a1 + mi*a0 + mj*a3 - mk*a2;
  float tj = mr*a2 - mi*a3 + mj*a0 + mk*a1;
  float tk = mr*a3 + mi*a2 - mj*a1 + mk*a0;
  float* o = T + (long long)dst * KDIM + h;
  atomicAdd(o,       tr);
  atomicAdd(o +  64, ti);
  atomicAdd(o + 128, tj);
  atomicAdd(o + 192, tk);
}

// ---------------------------------------------------------------------------
// H = relu(T_cat @ W_big + bias) via v_wmma_f32_16x16x32_f16.
// Block = 256 threads = 8 waves; block computes a 16x256 output strip.
// A strip staged in LDS as f16; each wave owns two 16x16 C tiles; K = 8x32.
// ---------------------------------------------------------------------------
__global__ __launch_bounds__(256) void qgemm_wmma(const float* __restrict__ T,
                                                  const _Float16* __restrict__ Wswz,
                                                  const float* __restrict__ bias,
                                                  float* __restrict__ H) {
  __shared__ _Float16 ldsA[16 * KDIM];            // 8 KB
  const int tid = threadIdx.x;
  const int m0  = blockIdx.x * 16;

  // cooperative load of the 16x256 A strip, f32 -> f16
#pragma unroll
  for (int i = 0; i < 16; ++i) {
    int idx = tid + i * 256;
    int r = idx >> 8, c = idx & 255;
    int gr = m0 + r;
    float v = (gr < N_NODES) ? T[(long long)gr * KDIM + c] : 0.0f;
    ldsA[r * KDIM + c] = (_Float16)v;
  }
  __syncthreads();

  const int wave = tid >> 5;
  const int lane = tid & 31;
  const int row  = lane & 15;
  const int koff = (lane >> 4) << 4;              // 0 or 16
  const int nt0  = wave * 2;

  v8f acc0 = {};
  v8f acc1 = {};

#pragma unroll
  for (int kc = 0; kc < 8; ++kc) {
    // A fragment: two 16-byte LDS reads per lane (ds_load_b128)
    const _Float16* ap = &ldsA[row * KDIM + kc * 32 + koff];
    v8h lo = *(const v8h*)ap;
    v8h hi = *(const v8h*)(ap + 8);
    v16h afrag;
#pragma unroll
    for (int i = 0; i < 8; ++i) { afrag[i] = lo[i]; afrag[8 + i] = hi[i]; }

    // B fragments: contiguous 32 B per lane (pre-swizzled)
    const _Float16* bbase = Wswz + ((size_t)(kc * 16 + nt0) * 32 + lane) * 16;
    if (kc < 7)
      __builtin_prefetch(Wswz + ((size_t)((kc + 1) * 16 + nt0) * 32 + lane) * 16, 0, 0);
    v16h b0 = *(const v16h*)bbase;
    v16h b1 = *(const v16h*)(bbase + 32 * 16);

    acc0 = __builtin_amdgcn_wmma_f32_16x16x32_f16(false, afrag, false, b0,
                                                  (short)0, acc0, false, false);
    acc1 = __builtin_amdgcn_wmma_f32_16x16x32_f16(false, afrag, false, b1,
                                                  (short)0, acc1, false, false);
  }

  // epilogue: bias + ReLU, C layout: col = lane&15, row = v (+8 for hi lanes)
  int col0 = nt0 * 16 + (lane & 15);
  int col1 = col0 + 16;
  float bs0 = bias[col0], bs1 = bias[col1];
#pragma unroll
  for (int v = 0; v < 8; ++v) {
    int r  = (lane < 16) ? v : (v + 8);
    int gr = m0 + r;
    if (gr < N_NODES) {
      float x0 = acc0[v] + bs0; x0 = x0 > 0.f ? x0 : 0.f;
      float x1 = acc1[v] + bs1; x1 = x1 > 0.f ? x1 : 0.f;
      H[(long long)gr * KDIM + col0] = x0;
      H[(long long)gr * KDIM + col1] = x1;
    }
  }
}

// ---------------------------------------------------------------------------
// Query head: feats(512) . Cw^T (2x512) + Cb, then log_softmax over 2 logits.
// One wave32 per query; feats order matches H layout: c-major, then (q0,q1).
// ---------------------------------------------------------------------------
__global__ void query_logits(const float* __restrict__ H,
                             const int* __restrict__ qe,
                             const float* __restrict__ Cw,
                             const float* __restrict__ Cb,
                             float* __restrict__ out) {
  int wave = threadIdx.x >> 5;
  int lane = threadIdx.x & 31;
  int q = blockIdx.x * 8 + wave;
  if (q >= N_QUERY) return;
  int q0 = qe[q * 2], q1 = qe[q * 2 + 1];
  float d0 = 0.f, d1 = 0.f;
#pragma unroll
  for (int j = 0; j < 16; ++j) {
    int f = lane + (j << 5);
    int node = ((f >> 6) & 1) ? q1 : q0;
    int c = f >> 7;
    int h = f & 63;
    float v = H[(long long)node * KDIM + c * 64 + h];
    d0 += v * Cw[f];
    d1 += v * Cw[512 + f];
  }
#pragma unroll
  for (int off = 16; off >= 1; off >>= 1) {
    d0 += __shfl_xor(d0, off, 32);
    d1 += __shfl_xor(d1, off, 32);
  }
  if (lane == 0) {
    d0 += Cb[0]; d1 += Cb[1];
    float m   = fmaxf(d0, d1);
    float lse = m + logf(expf(d0 - m) + expf(d1 - m));
    out[q * 2]     = d0 - lse;
    out[q * 2 + 1] = d1 - lse;
  }
}

// ---------------------------------------------------------------------------
extern "C" void kernel_launch(void* const* d_in, const int* in_sizes, int n_in,
                              void* d_out, int out_size, void* d_ws, size_t ws_size,
                              hipStream_t stream) {
  const float* Xr = (const float*)d_in[0];
  const float* Xi = (const float*)d_in[1];
  const float* Xj = (const float*)d_in[2];
  const float* Xk = (const float*)d_in[3];
  const float* nr = (const float*)d_in[4];
  const float* ni = (const float*)d_in[5];
  const float* nj = (const float*)d_in[6];
  const float* nk = (const float*)d_in[7];
  const int*   ei = (const int*)d_in[8];
  const int*   qe = (const int*)d_in[9];
  const float* W1 = (const float*)d_in[10];
  const float* b1 = (const float*)d_in[11];   // (4,64) flat == 256 concat order
  const float* W2 = (const float*)d_in[12];
  const float* b2 = (const float*)d_in[13];
  const float* Cw = (const float*)d_in[14];
  const float* Cb = (const float*)d_in[15];
  float* out = (float*)d_out;

  char* ws = (char*)d_ws;
  const size_t bytesT = (size_t)N_NODES * KDIM * sizeof(float);   // 102.4 MB
  float*     T     = (float*)ws;
  float*     H     = (float*)(ws + bytesT);
  _Float16*  Wswz1 = (_Float16*)(ws + 2 * bytesT);
  _Float16*  Wswz2 = Wswz1 + 65536;

  build_wbig<<<256, 256, 0, stream>>>(W1, Wswz1);
  build_wbig<<<256, 256, 0, stream>>>(W2, Wswz2);

  const int n4 = (N_NODES * KDIM) / 4;
  const int zgrid = (n4 + 255) / 256;
  const int ggrid = (N_NODES + 15) / 16;

  // Layer 1
  zero_f32<<<zgrid, 256, 0, stream>>>(T, n4);
  edge_scatter<<<N_EDGES / 4, 256, 0, stream>>>(Xr, Xi, Xj, Xk, 64,
                                                ei, nr, ni, nj, nk, T);
  qgemm_wmma<<<ggrid, 256, 0, stream>>>(T, Wswz1, b1, H);

  // Layer 2 (components live at H + a*64 with row stride 256)
  zero_f32<<<zgrid, 256, 0, stream>>>(T, n4);
  edge_scatter<<<N_EDGES / 4, 256, 0, stream>>>(H, H + 64, H + 128, H + 192, KDIM,
                                                ei, nr, ni, nj, nk, T);
  qgemm_wmma<<<ggrid, 256, 0, stream>>>(T, Wswz2, b2, H);

  // Link-prediction head
  query_logits<<<(N_QUERY + 7) / 8, 256, 0, stream>>>(H, qe, Cw, Cb, out);
}
